// S4Model_17901423690075
// MI455X (gfx1250) — matrix operand order repaired and checked by reading, not compiled, not run on hardware
//
#include <hip/hip_runtime.h>
#include <hip/hip_bf16.h>

// ---------------- problem constants (match reference) ----------------
#define B_   16
#define L_   2048
#define DIN_ 512
#define H_   512
#define N2_  32
#define NL_  4

typedef __attribute__((ext_vector_type(16))) __bf16 v16bf;
typedef __attribute__((ext_vector_type(8)))  float  v8f;

__device__ __forceinline__ unsigned short f2bf(float f) {
    unsigned u = __float_as_uint(f);
    unsigned r = (u + 0x7FFFu + ((u >> 16) & 1u)) >> 16;   // RNE
    return (unsigned short)r;
}

__device__ __forceinline__ unsigned lds_off32(const void* p) {
    return (unsigned)(reinterpret_cast<uintptr_t>(p));     // low 32 bits = LDS offset
}

// ---------------- fp32 -> bf16 weight conversion ----------------
__global__ void convert_kernel(const float* __restrict__ src,
                               unsigned short* __restrict__ dst, int n) {
    int i = blockIdx.x * 256 + threadIdx.x;
    if (i < n) dst[i] = f2bf(src[i]);
}

// ---------------- WMMA bf16 GEMM:  C[b] = act(W * Bact[b] (+bias)) ----------------
// W: MxK bf16 row-major.  Bact: per-batch KxN (fp32 or bf16, ld = N).  C: per-batch MxN fp32.
// EPI: 0 = ReLU, no bias.   1 = +bias, no activation.
// QUANT: quantize fp32 Bact to steps of 1/64 while staging (encoder input).
// BBF16: Bact is bf16 (ushort) -> copy-only staging.
#define BM 128
#define BN 128
#define BKK 32
#define LDSS 36   // padded k-stride (ushorts): 72B rows -> 8B aligned, conflict-free frag reads

union Frag { v16bf v; unsigned u[8]; };

template<int EPI, bool QUANT, bool BBF16>
__global__ __launch_bounds__(256)
void gemm_kernel(const unsigned short* __restrict__ Wq,
                 const void* __restrict__ Bact_,
                 float* __restrict__ Cout,
                 const float* __restrict__ bias,
                 int M, int K, int N)
{
    __shared__ unsigned short Alds[2][BM * LDSS];
    __shared__ unsigned short Blds[2][BN * LDSS];   // transposed: [n][k]

    const int tid  = threadIdx.x;
    const int lane = tid & 31;
    const int wid  = tid >> 5;
    const int wm   = wid >> 2;      // 0..1 -> 64-row slab
    const int wn   = wid & 3;       // 0..3 -> 32-col slab
    const int m0   = blockIdx.y * BM;
    const int n0   = blockIdx.x * BN;
    const int bb   = blockIdx.z;

    const float*          Bf = (const float*)Bact_ + (size_t)bb * K * N;
    const unsigned short* Bh = (const unsigned short*)Bact_ + (size_t)bb * K * N;
    float*                Cb = Cout + (size_t)bb * M * N;

    v8f acc[4][2] = {};
    const int half = lane >> 4;
    const int mr   = lane & 15;

    // ---- staging helpers ----
    auto stageA = [&](int k0, int buf) {   // async DMA: bf16 weights, row-major copy
        unsigned lbase = lds_off32(&Alds[buf][0]);
        #pragma unroll
        for (int i = 0; i < 4; ++i) {
            int e = tid + i * 256;            // 1024 x 8-byte chunks (128 rows x 64B)
            int r = e >> 3, c = e & 7;
            unsigned goff = (unsigned)((((m0 + r) * K + k0) << 1) + (c << 3));
            unsigned loff = lbase + r * (LDSS * 2) + (c << 3);
            asm volatile("global_load_async_to_lds_b64 %0, %1, %2"
                         :: "v"(loff), "v"(goff),
                            "s"((unsigned long long)(uintptr_t)Wq)
                         : "memory");
        }
    };
    auto stageB = [&](int k0, int buf) {   // transpose into [n][k]
        if (BBF16) {
            #pragma unroll
            for (int i = 0; i < 8; ++i) {
                int e = tid + i * 256;        // 2048 x u32 (2 ushorts along n)
                int k = e >> 6, n = (e & 63) << 1;
                unsigned pr = *reinterpret_cast<const unsigned*>(
                    &Bh[(size_t)(k0 + k) * N + n0 + n]);
                Blds[buf][n * LDSS + k]       = (unsigned short)(pr & 0xFFFFu);
                Blds[buf][(n + 1) * LDSS + k] = (unsigned short)(pr >> 16);
            }
        } else {
            #pragma unroll
            for (int i = 0; i < 16; ++i) {
                int e = tid + i * 256;
                int k = e >> 7, n = e & 127;
                float f = Bf[(size_t)(k0 + k) * N + n0 + n];
                if (QUANT) f = rintf(f * 64.0f) * 0.015625f;
                Blds[buf][n * LDSS + k] = f2bf(f);
            }
        }
    };

    const int nk = K / BKK;
    stageA(0, 0);
    stageB(0, 0);

    for (int ks = 0; ks < nk; ++ks) {
        const int cur = ks & 1;
        asm volatile("s_wait_asynccnt 0x0" ::: "memory");   // A-tile DMA done
        __syncthreads();                                    // B-tile ds stores done (all waves)

        if (ks + 1 < nk) {                                  // prefetch next slab into other buffer
            stageA((ks + 1) * BKK, cur ^ 1);
            stageB((ks + 1) * BKK, cur ^ 1);
        }

        // ---- fragments per ISA VGPR layouts ----
        Frag af[4];
        #pragma unroll
        for (int ti = 0; ti < 4; ++ti) {
            int rowo = (wm * 64 + ti * 16 + mr) * LDSS;
            #pragma unroll
            for (int v = 0; v < 8; ++v) {
                int kb = ((v & 3) << 1) + ((v >> 2) << 4) + (half << 3);
                af[ti].u[v] = *reinterpret_cast<const unsigned*>(&Alds[cur][rowo + kb]);
            }
        }
        Frag bfr[2];
        #pragma unroll
        for (int tj = 0; tj < 2; ++tj) {
            int colo = (wn * 32 + tj * 16 + mr) * LDSS;
            #pragma unroll
            for (int v = 0; v < 8; ++v) {
                int kb = ((v & 3) << 1) + ((v >> 2) << 4) + (half << 3);
                bfr[tj].u[v] = *reinterpret_cast<const unsigned*>(&Blds[cur][colo + kb]);
            }
        }
        #pragma unroll
        for (int ti = 0; ti < 4; ++ti)
            #pragma unroll
            for (int tj = 0; tj < 2; ++tj)
                acc[ti][tj] = __builtin_amdgcn_wmma_f32_16x16x32_bf16(
                    false, af[ti].v, false, bfr[tj].v,
                    (short)0, acc[ti][tj], false, false);
        __syncthreads();   // all waves done reading buf[cur] before it is restaged
    }

    // ---- epilogue ----
    #pragma unroll
    for (int ti = 0; ti < 4; ++ti) {
        #pragma unroll
        for (int tj = 0; tj < 2; ++tj) {
            int col = n0 + wn * 32 + tj * 16 + mr;
            #pragma unroll
            for (int r = 0; r < 8; ++r) {
                int row = m0 + wm * 64 + ti * 16 + r + half * 8;
                float vv = acc[ti][tj][r];
                if (EPI == 1) vv += bias[row];
                if (EPI == 0) vv = fmaxf(vv, 0.0f);
                Cb[(size_t)row * N + col] = vv;
            }
        }
    }
}

// ---------------- S4D diagonal-SSM scan (FFT-conv equivalent) + skip + GELU ----------------
// One wave32 per (b, channel); lane = mode n (N2 == 32).  Writes y in bf16.
__global__ __launch_bounds__(256)
void s4d_scan_kernel(const float* __restrict__ h, unsigned short* __restrict__ y,
                     const float* __restrict__ log_dt,
                     const float* __restrict__ log_A_real,
                     const float* __restrict__ A_imag,
                     const float* __restrict__ C_re,
                     const float* __restrict__ C_im,
                     const float* __restrict__ Dp,
                     int layer)
{
    const int wave = (blockIdx.x * blockDim.x + threadIdx.x) >> 5;   // b*H + ch
    const int lane = threadIdx.x & 31;
    const int ch   = wave & (H_ - 1);
    const int pidx = layer * H_ + ch;
    const int nidx = pidx * N2_ + lane;

    const float dt   = __expf(log_dt[pidx]);
    const float a_re = -__expf(log_A_real[nidx]);
    const float a_im = A_imag[nidx];
    const float zr = dt * a_re, zi = dt * a_im;
    float sn, cs;
    __sincosf(zi, &sn, &cs);
    const float ez   = __expf(zr);
    const float r_re = ez * cs, r_im = ez * sn;        // r = exp(dt*A)
    const float e1r = r_re - 1.0f, e1i = r_im;         // expm1(dt*A)
    const float inv = 1.0f / (a_re * a_re + a_im * a_im);
    const float qr = (e1r * a_re + e1i * a_im) * inv;  // expm1(dtA)/A
    const float qi = (e1i * a_re - e1r * a_im) * inv;
    const float cr = C_re[nidx], ci = C_im[nidx];
    const float Cc_re = cr * qr - ci * qi;
    const float Cc_im = cr * qi + ci * qr;
    const float Dv = Dp[pidx];

    const float*    hp = h + (size_t)wave * L_;
    unsigned short* yp = y + (size_t)wave * L_;

    float sr = 0.0f, si = 0.0f;
    for (int t0 = 0; t0 < L_; t0 += 32) {
        float uvec  = hp[t0 + lane];   // coalesced 32 inputs
        float yslot = 0.0f;
        for (int j = 0; j < 32; ++j) {
            float u = __shfl(uvec, j, 32);
            float nsr = fmaf(r_re, sr, fmaf(-r_im, si, u));
            float nsi = fmaf(r_re, si, r_im * sr);
            sr = nsr; si = nsi;
            float contrib = Cc_re * sr - Cc_im * si;
            #pragma unroll
            for (int d = 16; d > 0; d >>= 1)
                contrib += __shfl_xor(contrib, d, 32);
            float yv = 2.0f * contrib + Dv * u;
            if (lane == j) yslot = yv;
        }
        // exact GELU on the 32 gathered outputs, store as bf16 (GEMM input precision)
        float g = 0.5f * yslot * (1.0f + erff(yslot * 0.70710678118f));
        yp[t0 + lane] = f2bf(g);       // coalesced store
    }
}

// ---------------- GLU + residual:  h += z[:H] * sigmoid(z[H:]) ----------------
__global__ __launch_bounds__(256)
void glu_res_kernel(const float* __restrict__ z, float* __restrict__ h)
{
    size_t i = (size_t)blockIdx.x * 256 + threadIdx.x;   // over B*H*L
    size_t t = i % L_;
    size_t r = i / L_;
    int d = (int)(r % H_);
    int b = (int)(r / H_);
    size_t zb = ((size_t)b * 2 * H_) * L_;
    float z1 = z[zb + (size_t)d * L_ + t];
    float z2 = z[zb + (size_t)(d + H_) * L_ + t];
    float s  = 1.0f / (1.0f + __expf(-z2));
    h[i] += z1 * s;
}

// ---------------- channel LayerNorm (in place over h) ----------------
__global__ __launch_bounds__(256)
void layernorm_kernel(float* __restrict__ h,
                      const float* __restrict__ g,
                      const float* __restrict__ bb, int layer)
{
    int t = blockIdx.x * 256 + threadIdx.x;
    int b = blockIdx.y;
    float* hb = h + (size_t)b * H_ * L_;
    float s = 0.0f, s2 = 0.0f;
    for (int c = 0; c < H_; ++c) {
        float v = hb[(size_t)c * L_ + t];
        s += v; s2 += v * v;
    }
    const float mu  = s * (1.0f / H_);
    const float var = s2 * (1.0f / H_) - mu * mu;
    const float ri  = rsqrtf(var + 1e-5f);
    for (int c = 0; c < H_; ++c) {
        size_t idx = (size_t)c * L_ + t;
        float v = hb[idx];
        hb[idx] = (v - mu) * ri * g[layer * H_ + c] + bb[layer * H_ + c];
    }
}

// ---------------- host orchestration ----------------
extern "C" void kernel_launch(void* const* d_in, const int* in_sizes, int n_in,
                              void* d_out, int out_size, void* d_ws, size_t ws_size,
                              hipStream_t stream)
{
    (void)in_sizes; (void)n_in; (void)out_size; (void)ws_size;
    const float* x          = (const float*)d_in[0];
    const float* W_enc      = (const float*)d_in[1];
    const float* W_dec      = (const float*)d_in[2];
    const float* log_dt     = (const float*)d_in[3];
    const float* log_A_real = (const float*)d_in[4];
    const float* A_imag     = (const float*)d_in[5];
    const float* C_re       = (const float*)d_in[6];
    const float* C_im       = (const float*)d_in[7];
    const float* Dp         = (const float*)d_in[8];
    const float* W_out      = (const float*)d_in[9];
    const float* b_out      = (const float*)d_in[10];
    const float* ln_g       = (const float*)d_in[11];
    const float* ln_b       = (const float*)d_in[12];

    char* p = (char*)d_ws;
    float*          h = (float*)p;          p += (size_t)B_ * H_ * L_ * sizeof(float);     // 64 MB
    unsigned short* y = (unsigned short*)p; p += (size_t)B_ * H_ * L_ * 2;                 // 32 MB (bf16)
    float*          z = (float*)p;          p += (size_t)B_ * 2 * H_ * L_ * sizeof(float); // 128 MB
    unsigned short* wenc = (unsigned short*)p; p += (size_t)H_ * DIN_ * 2;
    unsigned short* wdec = (unsigned short*)p; p += (size_t)DIN_ * H_ * 2;
    unsigned short* wout = (unsigned short*)p; p += (size_t)NL_ * 2 * H_ * H_ * 2;

    // one-time (per launch) weight conversion to bf16
    convert_kernel<<<(H_ * DIN_ + 255) / 256, 256, 0, stream>>>(W_enc, wenc, H_ * DIN_);
    convert_kernel<<<(DIN_ * H_ + 255) / 256, 256, 0, stream>>>(W_dec, wdec, DIN_ * H_);
    convert_kernel<<<(NL_ * 2 * H_ * H_ + 255) / 256, 256, 0, stream>>>(W_out, wout, NL_ * 2 * H_ * H_);

    dim3 blk(256);

    // encoder: h = relu(W_enc @ quantize(x))   (fp32 B, fused quantize)
    dim3 genc(L_ / BN, H_ / BM, B_);
    gemm_kernel<0, true, false><<<genc, blk, 0, stream>>>(wenc, x, h, nullptr, H_, DIN_, L_);

    for (int layer = 0; layer < NL_; ++layer) {
        // SSM scan + skip + GELU -> y (bf16)
        s4d_scan_kernel<<<(B_ * H_) / 8, 256, 0, stream>>>(
            h, y, log_dt, log_A_real, A_imag, C_re, C_im, Dp, layer);
        // z = W_out[layer] @ y + b_out[layer]   (bf16 B path, copy-only staging)
        dim3 gout(L_ / BN, (2 * H_) / BM, B_);
        gemm_kernel<1, false, true><<<gout, blk, 0, stream>>>(
            wout + (size_t)layer * 2 * H_ * H_, y, z, b_out + layer * 2 * H_, 2 * H_, H_, L_);
        // GLU + residual, then LayerNorm over channels (in place)
        glu_res_kernel<<<(unsigned)((size_t)B_ * H_ * L_ / 256), 256, 0, stream>>>(z, h);
        layernorm_kernel<<<dim3(L_ / 256, B_), 256, 0, stream>>>(h, ln_g, ln_b, layer);
    }

    // decoder: out = relu(W_dec @ h)   (fp32 B)
    dim3 gdec(L_ / BN, DIN_ / BM, B_);
    gemm_kernel<0, false, false><<<gdec, blk, 0, stream>>>(wdec, h, (float*)d_out, nullptr, DIN_, H_, L_);
}
